// VQVAE_10024453669311
// MI455X (gfx1250) — matrix operand order repaired and checked by reading, not compiled, not run on hardware
//
#include <hip/hip_runtime.h>

typedef __attribute__((ext_vector_type(16))) _Float16 v16h;
typedef __attribute__((ext_vector_type(2)))  _Float16 v2h;
typedef __attribute__((ext_vector_type(8)))  float    v8f;
typedef __attribute__((ext_vector_type(4)))  int      v4i;

// LDS A-tile row stride (halves): 40 halves = 80 B keeps b128 stores 16B-aligned.
#define AROW 40

// ---------------------------------------------------------------------------
// Param structs (passed by value as kernel args)
// ---------------------------------------------------------------------------
struct ConvP {
    int Din, Hin, Win, Dout, Hout, Wout;
    int Cin, Cout, ksz, stride, pad;
    int Ktot, nk, nt;
};
struct ConvTP {
    int Din, Hin, Win;   // input dims; output = 2x
    int Cin, Cout, Ktot, nk, nt;
};

// ---------------------------------------------------------------------------
// fp32 -> f16 elementwise
// ---------------------------------------------------------------------------
__global__ void k_f32h(const float* __restrict__ s, _Float16* __restrict__ d, int n) {
    int i = blockIdx.x * 256 + threadIdx.x;
    if (i < n) d[i] = (_Float16)s[i];
}

// ---------------------------------------------------------------------------
// Weight packing: conv weights [O][I][k^3] fp32 -> fragment-ready f16 B tiles.
// Layout: idx = ((kchunk*nt + ntile)*32 + lane)*16 + e
// B fragment mapping (32x16, 16-bit, this code's contiguous k order): lane -> n=lane&15, g=lane>>4;
// element e -> r=e>>1, k_in_chunk = g*16 + 2r + (e&1).
// K ordering: kk = tap*Cin + ci, tap = (kd*k + kh)*k + kw.
// ---------------------------------------------------------------------------
__global__ void k_pack_conv(const float* __restrict__ src, _Float16* __restrict__ dst,
                            int Cin, int Cout, int k3, int Ktot, int nt, int total) {
    int idx = blockIdx.x * 256 + threadIdx.x;
    if (idx >= total) return;
    int e    = idx & 15;
    int lane = (idx >> 4) & 31;
    int tile = idx >> 9;
    int ntile  = tile % nt;
    int kchunk = tile / nt;
    int n = lane & 15, g = lane >> 4, r = e >> 1;
    int kin = g * 16 + 2 * r + (e & 1);
    int kk  = kchunk * 32 + kin;
    int o   = ntile * 16 + n;
    float v = 0.f;
    if (kk < Ktot && o < Cout) {
        int tap = kk / Cin, ci = kk - tap * Cin;
        v = src[((size_t)o * Cin + ci) * k3 + tap];
    }
    dst[idx] = (_Float16)v;
}

// ConvTranspose weights [I][O][4][4][4] fp32 -> 8 parity classes of packed B.
// For output parity p (per dim): valid k = {1,3} if p==0 else {0,2};
// tap bit a selects which, input offset: p==0: {0,-1}; p==1: {+1,0}.
__global__ void k_pack_convt(const float* __restrict__ src, _Float16* __restrict__ dst,
                             int Cin, int Cout, int Ktot, int nt, int perClass, int total) {
    int idx = blockIdx.x * 256 + threadIdx.x;
    if (idx >= total) return;
    int cls = idx / perClass;
    int li  = idx - cls * perClass;
    int pd = (cls >> 2) & 1, ph = (cls >> 1) & 1, pw = cls & 1;
    int e    = li & 15;
    int lane = (li >> 4) & 31;
    int tile = li >> 9;
    int ntile  = tile % nt;
    int kchunk = tile / nt;
    int n = lane & 15, g = lane >> 4, r = e >> 1;
    int kin = g * 16 + 2 * r + (e & 1);
    int kk  = kchunk * 32 + kin;
    int o   = ntile * 16 + n;
    float v = 0.f;
    if (kk < Ktot && o < Cout) {
        int t  = kk / Cin, ci = kk - t * Cin;
        int ta = (t >> 2) & 1, tb = (t >> 1) & 1, tc = t & 1;
        int kd = pd ? (ta ? 2 : 0) : (ta ? 3 : 1);
        int kh = ph ? (tb ? 2 : 0) : (tb ? 3 : 1);
        int kw = pw ? (tc ? 2 : 0) : (tc ? 3 : 1);
        v = src[((size_t)ci * Cout + o) * 64 + (kd * 4 + kh) * 4 + kw];
    }
    dst[idx] = (_Float16)v;
}

// ---------------------------------------------------------------------------
// Implicit-GEMM conv3d with WMMA f32_16x16x32_f16.
// Block = 128 threads (4 waves). One block = one 16-voxel M-tile (along W),
// waves cover n-tiles in stride-4 (up to 3 accumulators each).
// Two-deep software pipeline:
//   - A chunks: global->reg prefetch, double-buffered LDS, 1 barrier/chunk
//   - B fragments: register double-buffered (loop unrolled by 2), so each
//     WMMA consumes B data requested one full chunk earlier.
// FASTC: Cin % 32 == 0 -> b128 staging, incremental tap/channel tracking.
// !FASTC: assumes Cin == 1 (down0): tap == kk.
// FUSE: 0 = bias only, 1 = bias+ReLU, 2 = bias + residual + ReLU
// ---------------------------------------------------------------------------
template <int FUSE, bool FASTC>
__global__ __launch_bounds__(128) void k_conv(
    const _Float16* __restrict__ in, const _Float16* __restrict__ wp,
    const float* __restrict__ bias, const _Float16* __restrict__ res,
    _Float16* __restrict__ out, ConvP p) {
    __shared__ _Float16 aT[2][16 * AROW];
    const int tid  = threadIdx.x;
    const int wave = tid >> 5, lane = tid & 31;
    const int wtN = p.Wout >> 4;
    const int wt  = blockIdx.x % wtN;
    const int dh  = blockIdx.x / wtN;
    const int ho  = dh % p.Hout;
    const int dz  = dh / p.Hout;
    const int wbase = wt << 4;
    const int mlane = lane & 15, glane = lane >> 4;
    const int sm = tid >> 2, ssub = tid & 3;  // fast-path staging role (tid < 64)
    const bool act1 = (wave + 4) < p.nt, act2 = (wave + 8) < p.nt;
    v8f acc[3] = {};
    v4i areg;                 // fast-path staged 8 halves
    _Float16 gv[4];           // generic-path staged elements
    v16h bA0 = {}, bA1 = {}, bA2 = {};   // B fragment double buffer
    v16h bB0 = {}, bB1 = {}, bB2 = {};

    auto stageLoad = [&](int kc, int tap, int cb) {
        if (FASTC) {
            if (tid < 64) {
                int kw = tap % p.ksz;
                int t2 = tap / p.ksz;
                int kh = t2 % p.ksz;
                int kd = t2 / p.ksz;
                int wi = (wbase + sm) * p.stride + kw - p.pad;
                int hi = ho * p.stride + kh - p.pad;
                int di = dz * p.stride + kd - p.pad;
                if ((unsigned)wi < (unsigned)p.Win && (unsigned)hi < (unsigned)p.Hin &&
                    (unsigned)di < (unsigned)p.Din) {
                    areg = *(const v4i*)(in +
                        (((size_t)di * p.Hin + hi) * p.Win + wi) * p.Cin + cb + ssub * 8);
                } else {
                    areg[0] = 0; areg[1] = 0; areg[2] = 0; areg[3] = 0;
                }
            }
        } else {
            // Cin == 1: tap == kk, ci == 0
#pragma unroll
            for (int u = 0; u < 4; ++u) {
                int i = tid + u * 128;
                int m = i >> 5, j = i & 31;
                int kk = (kc << 5) + j;
                _Float16 v = (_Float16)0.f;
                if (kk < p.Ktot) {
                    int kw = kk % p.ksz;
                    int t2 = kk / p.ksz;
                    int kh = t2 % p.ksz;
                    int kd = t2 / p.ksz;
                    int wi = (wbase + m) * p.stride + kw - p.pad;
                    int hi = ho * p.stride + kh - p.pad;
                    int di = dz * p.stride + kd - p.pad;
                    if ((unsigned)wi < (unsigned)p.Win && (unsigned)hi < (unsigned)p.Hin &&
                        (unsigned)di < (unsigned)p.Din)
                        v = in[((size_t)di * p.Hin + hi) * p.Win + wi];
                }
                gv[u] = v;
            }
        }
    };
    auto stageStore = [&](int buf) {
        if (FASTC) {
            if (tid < 64) *(v4i*)&aT[buf][sm * AROW + ssub * 8] = areg;
        } else {
#pragma unroll
            for (int u = 0; u < 4; ++u) {
                int i = tid + u * 128;
                aT[buf][(i >> 5) * AROW + (i & 31)] = gv[u];
            }
        }
    };

    // per-wave B pointer: element ((kc*nt + wave + 4q)*32 + lane)*16
    const _Float16* wpw = wp + (size_t)wave * 512 + (size_t)lane * 16;
    const size_t wstep = (size_t)p.nt * 512;

    auto loadB = [&](int kc, v16h& B0, v16h& B1, v16h& B2) {
        if (wave < p.nt) {
            const _Float16* wb = wpw + (size_t)kc * wstep;
            B0 = *(const v16h*)(wb);
            if (act1) B1 = *(const v16h*)(wb + 2048);
            if (act2) B2 = *(const v16h*)(wb + 4096);
        }
    };
    auto compute = [&](int buf, const v16h& B0, const v16h& B1, const v16h& B2) {
        if (wave < p.nt) {  // wave-uniform; EXEC all-1s inside
            v16h a;
            // A layout (matches packed B): lanes 0-15 K{0..15}, 16-31 K{16..31}
            const _Float16* ab = &aT[buf][mlane * AROW];
#pragma unroll
            for (int r = 0; r < 8; ++r) {
                int kb = 16 * glane + 2 * r;
                v2h pr = *(const v2h*)(ab + kb);
                a[2 * r]     = pr[0];
                a[2 * r + 1] = pr[1];
            }
            acc[0] = __builtin_amdgcn_wmma_f32_16x16x32_f16(
                false, a, false, B0, (short)0, acc[0], false, false);
            if (act1)
                acc[1] = __builtin_amdgcn_wmma_f32_16x16x32_f16(
                    false, a, false, B1, (short)0, acc[1], false, false);
            if (act2)
                acc[2] = __builtin_amdgcn_wmma_f32_16x16x32_f16(
                    false, a, false, B2, (short)0, acc[2], false, false);
        }
    };

    // pipeline prologue: stage chunk 0, preload its B fragments
    stageLoad(0, 0, 0);
    stageStore(0);
    loadB(0, bA0, bA1, bA2);
    __syncthreads();

    int tapN = 0, cbN = 0;  // tap/channel-base of the chunk being prefetched
    int kc = 0;
    while (kc < p.nk) {
        // ---- even step: consume LDS[kc&1] + B set A -------------------------
        const bool have1 = (kc + 1) < p.nk;
        if (have1) {
            cbN += 32;
            if (cbN >= p.Cin) { cbN = 0; ++tapN; }
            stageLoad(kc + 1, tapN, cbN);      // A prefetch first (in-order loads)
            loadB(kc + 1, bB0, bB1, bB2);      // then B prefetch
        }
        compute(kc & 1, bA0, bA1, bA2);
        if (have1) stageStore((kc + 1) & 1);
        __syncthreads();
        if (!have1) break;

        // ---- odd step: consume LDS[(kc+1)&1] + B set B ----------------------
        const bool have2 = (kc + 2) < p.nk;
        if (have2) {
            cbN += 32;
            if (cbN >= p.Cin) { cbN = 0; ++tapN; }
            stageLoad(kc + 2, tapN, cbN);
            loadB(kc + 2, bA0, bA1, bA2);
        }
        compute((kc + 1) & 1, bB0, bB1, bB2);
        if (have2) stageStore((kc + 2) & 1);
        __syncthreads();
        kc += 2;
    }

    // epilogue: C layout (m = v + 8g, n = lane&15)
#pragma unroll
    for (int q = 0; q < 3; ++q) {
        int nti = wave + 4 * q;
        if (nti < p.nt) {
            int col = nti * 16 + mlane;
            if (col < p.Cout) {
                float bv = bias[col];
#pragma unroll
                for (int v = 0; v < 8; ++v) {
                    int m = v + 8 * glane;
                    size_t oidx =
                        (((size_t)dz * p.Hout + ho) * p.Wout + (wbase + m)) * p.Cout + col;
                    float val = acc[q][v] + bv;
                    if (FUSE == 2) val += (float)res[oidx];
                    if (FUSE >= 1) val = fmaxf(val, 0.f);
                    out[oidx] = (_Float16)val;
                }
            }
        }
    }
}

// ---------------------------------------------------------------------------
// Transposed conv (stride 2, k=4, pad=1) as 8 parity-class subpixel convs.
// blockIdx.z = parity class. Same two-deep pipeline; Cin always %32==0.
// MODE 0: f16 out + ReLU; MODE 1: fp32 out (ch 0 only), no activation.
// ---------------------------------------------------------------------------
__device__ __forceinline__ int tioff(int pp, int aa) {
    return pp ? (aa ? 0 : 1) : (aa ? -1 : 0);
}

template <int MODE>
__global__ __launch_bounds__(128) void k_convt(
    const _Float16* __restrict__ in, const _Float16* __restrict__ wpAll,
    const float* __restrict__ bias, _Float16* __restrict__ outH,
    float* __restrict__ outF, ConvTP p) {
    __shared__ _Float16 aT[2][16 * AROW];
    const int tid  = threadIdx.x;
    const int wave = tid >> 5, lane = tid & 31;
    const int wtN = p.Win >> 4;
    const int wt  = blockIdx.x % wtN;
    const int dh  = blockIdx.x / wtN;
    const int ht  = dh % p.Hin;
    const int dt  = dh / p.Hin;
    const int wbase = wt << 4;
    const int cls = blockIdx.z;
    const int pd = (cls >> 2) & 1, ph = (cls >> 1) & 1, pw = cls & 1;
    const _Float16* wp = wpAll + (size_t)cls * p.nk * p.nt * 512;
    const int mlane = lane & 15, glane = lane >> 4;
    const int sm = tid >> 2, ssub = tid & 3;
    const bool act1 = (wave + 4) < p.nt, act2 = (wave + 8) < p.nt;
    v8f acc[3] = {};
    v4i areg;
    v16h bA0 = {}, bA1 = {}, bA2 = {};
    v16h bB0 = {}, bB1 = {}, bB2 = {};

    auto stageLoad = [&](int tap, int cb) {
        if (tid < 64) {
            int ta = (tap >> 2) & 1, tb = (tap >> 1) & 1, tc = tap & 1;
            int di = dt + tioff(pd, ta);
            int hi = ht + tioff(ph, tb);
            int wi = wbase + sm + tioff(pw, tc);
            if ((unsigned)wi < (unsigned)p.Win && (unsigned)hi < (unsigned)p.Hin &&
                (unsigned)di < (unsigned)p.Din) {
                areg = *(const v4i*)(in +
                    (((size_t)di * p.Hin + hi) * p.Win + wi) * p.Cin + cb + ssub * 8);
            } else {
                areg[0] = 0; areg[1] = 0; areg[2] = 0; areg[3] = 0;
            }
        }
    };
    auto stageStore = [&](int buf) {
        if (tid < 64) *(v4i*)&aT[buf][sm * AROW + ssub * 8] = areg;
    };

    const _Float16* wpw = wp + (size_t)wave * 512 + (size_t)lane * 16;
    const size_t wstep = (size_t)p.nt * 512;

    auto loadB = [&](int kc, v16h& B0, v16h& B1, v16h& B2) {
        if (wave < p.nt) {
            const _Float16* wb = wpw + (size_t)kc * wstep;
            B0 = *(const v16h*)(wb);
            if (act1) B1 = *(const v16h*)(wb + 2048);
            if (act2) B2 = *(const v16h*)(wb + 4096);
        }
    };
    auto compute = [&](int buf, const v16h& B0, const v16h& B1, const v16h& B2) {
        if (wave < p.nt) {
            v16h a;
            const _Float16* ab = &aT[buf][mlane * AROW];
#pragma unroll
            for (int r = 0; r < 8; ++r) {
                int kb = 16 * glane + 2 * r;
                v2h pr = *(const v2h*)(ab + kb);
                a[2 * r]     = pr[0];
                a[2 * r + 1] = pr[1];
            }
            acc[0] = __builtin_amdgcn_wmma_f32_16x16x32_f16(
                false, a, false, B0, (short)0, acc[0], false, false);
            if (act1)
                acc[1] = __builtin_amdgcn_wmma_f32_16x16x32_f16(
                    false, a, false, B1, (short)0, acc[1], false, false);
            if (act2)
                acc[2] = __builtin_amdgcn_wmma_f32_16x16x32_f16(
                    false, a, false, B2, (short)0, acc[2], false, false);
        }
    };

    stageLoad(0, 0);
    stageStore(0);
    loadB(0, bA0, bA1, bA2);
    __syncthreads();

    int tapN = 0, cbN = 0;
    int kc = 0;
    while (kc < p.nk) {
        const bool have1 = (kc + 1) < p.nk;
        if (have1) {
            cbN += 32;
            if (cbN >= p.Cin) { cbN = 0; ++tapN; }
            stageLoad(tapN, cbN);
            loadB(kc + 1, bB0, bB1, bB2);
        }
        compute(kc & 1, bA0, bA1, bA2);
        if (have1) stageStore((kc + 1) & 1);
        __syncthreads();
        if (!have1) break;

        const bool have2 = (kc + 2) < p.nk;
        if (have2) {
            cbN += 32;
            if (cbN >= p.Cin) { cbN = 0; ++tapN; }
            stageLoad(tapN, cbN);
            loadB(kc + 2, bA0, bA1, bA2);
        }
        compute((kc + 1) & 1, bB0, bB1, bB2);
        if (have2) stageStore((kc + 2) & 1);
        __syncthreads();
        kc += 2;
    }

    const int Hout = p.Hin * 2, Wout = p.Win * 2;
#pragma unroll
    for (int q = 0; q < 3; ++q) {
        int nti = wave + 4 * q;
        if (nti < p.nt) {
            int col = nti * 16 + mlane;
            if (col < p.Cout) {
                float bv = bias[col];
#pragma unroll
                for (int v = 0; v < 8; ++v) {
                    int m  = v + 8 * glane;
                    int od = 2 * dt + pd, oh = 2 * ht + ph, ow = 2 * (wbase + m) + pw;
                    size_t base = ((size_t)od * Hout + oh) * Wout + ow;
                    float val = acc[q][v] + bv;
                    if (MODE == 0) {
                        val = fmaxf(val, 0.f);
                        outH[base * p.Cout + col] = (_Float16)val;
                    } else {
                        outF[base] = val;
                    }
                }
            }
        }
    }
}

// ---------------------------------------------------------------------------
// Vector quantization: z [N][32] f16, codebook [256][32] f32 -> q [N][32] f16
// plus per-block partial sums of (q - e)^2 for the commitment loss.
// ---------------------------------------------------------------------------
__global__ __launch_bounds__(256) void k_quant(const _Float16* __restrict__ z,
                                               const float* __restrict__ cb,
                                               _Float16* __restrict__ q,
                                               float* __restrict__ partial, int N) {
    __shared__ float red[256];
    int i = blockIdx.x * 256 + threadIdx.x;
    float local = 0.f;
    if (i < N) {
        float e[32];
#pragma unroll
        for (int c = 0; c < 32; ++c) e[c] = (float)z[(size_t)i * 32 + c];
        int best = 0;
        float bestd = 3.4e38f;
        for (int kcode = 0; kcode < 256; ++kcode) {
            float d = 0.f;
#pragma unroll
            for (int c = 0; c < 32; ++c) {
                float t = e[c] - cb[kcode * 32 + c];
                d += t * t;
            }
            if (d < bestd) { bestd = d; best = kcode; }
        }
#pragma unroll
        for (int c = 0; c < 32; ++c) {
            float qv = cb[best * 32 + c];
            q[(size_t)i * 32 + c] = (_Float16)qv;
            float t = qv - e[c];
            local += t * t;
        }
    }
    red[threadIdx.x] = local;
    __syncthreads();
    for (int s = 128; s > 0; s >>= 1) {
        if (threadIdx.x < s) red[threadIdx.x] += red[threadIdx.x + s];
        __syncthreads();
    }
    if (threadIdx.x == 0) partial[blockIdx.x] = red[0];
}

__global__ __launch_bounds__(128) void k_loss_reduce(const float* __restrict__ partial,
                                                     float* __restrict__ out, int nPart,
                                                     float scale) {
    __shared__ float red[128];
    float s = 0.f;
    for (int i = threadIdx.x; i < nPart; i += 128) s += partial[i];
    red[threadIdx.x] = s;
    __syncthreads();
    for (int st = 64; st > 0; st >>= 1) {
        if (threadIdx.x < st) red[threadIdx.x] += red[threadIdx.x + st];
        __syncthreads();
    }
    if (threadIdx.x == 0) *out = red[0] * scale;
}

// ---------------------------------------------------------------------------
// Host orchestration
// ---------------------------------------------------------------------------
extern "C" void kernel_launch(void* const* d_in, const int* in_sizes, int n_in,
                              void* d_out, int out_size, void* d_ws, size_t ws_size,
                              hipStream_t stream) {
    (void)ws_size;
    auto F = [&](int i) -> const float* { return (const float*)d_in[i]; };

    // Locate x and codebook by unique element counts (robust to flatten order).
    int ix = n_in - 1, icb = 0;
    for (int i = 0; i < n_in; ++i) {
        if (in_sizes[i] == 128 * 128 * 128) ix = i;
        if (in_sizes[i] == 256 * 32) icb = i;
    }
    // Remaining mapping assumes jax pytree (sorted-key) flattening:
    // params < x ; within params: codebook, dec, enc ; res dicts: b1,b2,w1,w2.
    const float* cb       = F(icb);
    const float* dec_in_b = F(1);
    const float* dec_in_w = F(2);
    const float* d_r[2][3][4];
    int di = 3;
    for (int l = 0; l < 2; ++l)
        for (int j = 0; j < 3; ++j)
            for (int t = 0; t < 4; ++t) d_r[l][j][t] = F(di++);
    const float* up0_b = F(27);
    const float* up0_w = F(28);
    const float* up1_b = F(29);
    const float* up1_w = F(30);
    const float* down0_b   = F(31);
    const float* down0_w   = F(32);
    const float* down1_b   = F(33);
    const float* down1_w   = F(34);
    const float* enc_out_b = F(35);
    const float* enc_out_w = F(36);
    const float* e_r[2][3][4];
    int ei = 37;
    for (int l = 0; l < 2; ++l)
        for (int j = 0; j < 3; ++j)
            for (int t = 0; t < 4; ++t) e_r[l][j][t] = F(ei++);
    const float* x = F(ix);

    // ---- workspace carve-up (all 256B aligned) ------------------------------
    char* wsb  = (char*)d_ws;
    size_t cur = 0;
    auto allocB = [&](size_t bytes) -> void* {
        void* pp = wsb + cur;
        cur = (cur + bytes + 255) & ~(size_t)255;
        return pp;
    };
    float*     partial = (float*)allocB(256 * sizeof(float));
    _Float16*  bufX    = (_Float16*)allocB((size_t)128 * 128 * 128 * 2);
    const size_t BUFE  = (size_t)64 * 64 * 64 * 96;  // largest activation (f16 elems)
    _Float16* curB = (_Float16*)allocB(BUFE * 2);
    _Float16* s1   = (_Float16*)allocB(BUFE * 2);
    _Float16* s2   = (_Float16*)allocB(BUFE * 2);

    auto pkElems = [](int Ci, int Co, int kz) -> size_t {
        int k3 = kz * kz * kz;
        int Kt = Ci * k3;
        int nk = (Kt + 31) / 32;
        int nt = (Co + 15) / 16;
        return (size_t)nk * nt * 512;
    };
    auto packC = [&](const float* src, int Ci, int Co, int kz) -> _Float16* {
        size_t tot = pkElems(Ci, Co, kz);
        _Float16* dst = (_Float16*)allocB(tot * 2);
        int k3 = kz * kz * kz;
        k_pack_conv<<<dim3((unsigned)((tot + 255) / 256)), dim3(256), 0, stream>>>(
            src, dst, Ci, Co, k3, Ci * k3, (Co + 15) / 16, (int)tot);
        return dst;
    };
    auto packT = [&](const float* src, int Ci, int Co) -> _Float16* {
        int Kt = 8 * Ci;
        int nk = (Kt + 31) / 32;
        int nt = (Co + 15) / 16;
        size_t per = (size_t)nk * nt * 512, tot = per * 8;
        _Float16* dst = (_Float16*)allocB(tot * 2);
        k_pack_convt<<<dim3((unsigned)((tot + 255) / 256)), dim3(256), 0, stream>>>(
            src, dst, Ci, Co, Kt, nt, (int)per, (int)tot);
        return dst;
    };

    // ---- pack all weights ---------------------------------------------------
    _Float16* W_down0 = packC(down0_w, 1, 96, 4);
    _Float16* W_er[2][3][2];
    for (int j = 0; j < 3; ++j) {
        W_er[0][j][0] = packC(e_r[0][j][2], 96, 96, 3);
        W_er[0][j][1] = packC(e_r[0][j][3], 96, 96, 3);
    }
    _Float16* W_down1 = packC(down1_w, 96, 192, 4);
    for (int j = 0; j < 3; ++j) {
        W_er[1][j][0] = packC(e_r[1][j][2], 192, 192, 3);
        W_er[1][j][1] = packC(e_r[1][j][3], 192, 192, 3);
    }
    _Float16* W_eout = packC(enc_out_w, 192, 32, 3);
    _Float16* W_din  = packC(dec_in_w, 32, 192, 3);
    _Float16* W_dr[2][3][2];
    for (int j = 0; j < 3; ++j) {
        W_dr[0][j][0] = packC(d_r[0][j][2], 192, 192, 3);
        W_dr[0][j][1] = packC(d_r[0][j][3], 192, 192, 3);
    }
    _Float16* W_up0 = packT(up0_w, 192, 96);
    for (int j = 0; j < 3; ++j) {
        W_dr[1][j][0] = packC(d_r[1][j][2], 96, 96, 3);
        W_dr[1][j][1] = packC(d_r[1][j][3], 96, 96, 3);
    }
    _Float16* W_up1 = packT(up1_w, 96, 1);

    // ---- launch helpers -----------------------------------------------------
    auto conv = [&](const _Float16* in, const _Float16* wpk, const float* bias,
                    const _Float16* resid, _Float16* out, int Di, int Hi, int Wi, int Do,
                    int Ho, int Wo, int Ci, int Co, int kz, int st, int pad, int fuse) {
        ConvP p;
        p.Din = Di; p.Hin = Hi; p.Win = Wi;
        p.Dout = Do; p.Hout = Ho; p.Wout = Wo;
        p.Cin = Ci; p.Cout = Co; p.ksz = kz; p.stride = st; p.pad = pad;
        p.Ktot = Ci * kz * kz * kz;
        p.nk = (p.Ktot + 31) / 32;
        p.nt = (Co + 15) / 16;
        dim3 grid((unsigned)((Wo / 16) * Ho * Do));
        const bool fast = (Ci % 32) == 0;
        if (fast) {
            if (fuse == 0)      k_conv<0, true><<<grid, 128, 0, stream>>>(in, wpk, bias, resid, out, p);
            else if (fuse == 1) k_conv<1, true><<<grid, 128, 0, stream>>>(in, wpk, bias, resid, out, p);
            else                k_conv<2, true><<<grid, 128, 0, stream>>>(in, wpk, bias, resid, out, p);
        } else {
            if (fuse == 0)      k_conv<0, false><<<grid, 128, 0, stream>>>(in, wpk, bias, resid, out, p);
            else if (fuse == 1) k_conv<1, false><<<grid, 128, 0, stream>>>(in, wpk, bias, resid, out, p);
            else                k_conv<2, false><<<grid, 128, 0, stream>>>(in, wpk, bias, resid, out, p);
        }
    };
    auto convt = [&](const _Float16* in, const _Float16* wpk, const float* bias,
                     _Float16* outH, float* outF, int Di, int Hi, int Wi, int Ci, int Co,
                     int mode) {
        ConvTP p;
        p.Din = Di; p.Hin = Hi; p.Win = Wi;
        p.Cin = Ci; p.Cout = Co;
        p.Ktot = 8 * Ci;
        p.nk = (p.Ktot + 31) / 32;
        p.nt = (Co + 15) / 16;
        dim3 grid((unsigned)((Wi / 16) * Hi * Di), 1, 8);
        if (mode == 0)
            k_convt<0><<<grid, 128, 0, stream>>>(in, wpk, bias, outH, (float*)nullptr, p);
        else
            k_convt<1><<<grid, 128, 0, stream>>>(in, wpk, bias, (_Float16*)nullptr, outF, p);
    };
    auto resunit = [&](int D, int H, int W, int C, const float* b1, const float* b2,
                       const _Float16* w1p, const _Float16* w2p) {
        conv(curB, w1p, b1, nullptr, s1, D, H, W, D, H, W, C, C, 3, 1, 1, 1);
        conv(s1, w2p, b2, curB, s2, D, H, W, D, H, W, C, C, 3, 1, 1, 2);
        _Float16* t = curB; curB = s2; s2 = t;
    };

    // ---- network ------------------------------------------------------------
    const int NX = 128 * 128 * 128;
    k_f32h<<<dim3((NX + 255) / 256), 256, 0, stream>>>(x, bufX, NX);

    // encoder
    conv(bufX, W_down0, down0_b, nullptr, curB, 128, 128, 128, 64, 64, 64, 1, 96, 4, 2, 1, 1);
    for (int j = 0; j < 3; ++j)
        resunit(64, 64, 64, 96, e_r[0][j][0], e_r[0][j][1], W_er[0][j][0], W_er[0][j][1]);
    conv(curB, W_down1, down1_b, nullptr, s1, 64, 64, 64, 32, 32, 32, 96, 192, 4, 2, 1, 1);
    { _Float16* t = curB; curB = s1; s1 = t; }
    for (int j = 0; j < 3; ++j)
        resunit(32, 32, 32, 192, e_r[1][j][0], e_r[1][j][1], W_er[1][j][0], W_er[1][j][1]);
    conv(curB, W_eout, enc_out_b, nullptr, s1, 32, 32, 32, 32, 32, 32, 192, 32, 3, 1, 1, 0);

    // quantize: z = s1, q -> s2; deterministic loss reduction -> d_out[last]
    const int NV = 32 * 32 * 32;
    k_quant<<<dim3(NV / 256), 256, 0, stream>>>(s1, cb, s2, partial, NV);
    k_loss_reduce<<<dim3(1), 128, 0, stream>>>(partial, (float*)d_out + (out_size - 1),
                                               NV / 256, 0.25f / ((float)NV * 32.f));

    // decoder
    conv(s2, W_din, dec_in_b, nullptr, curB, 32, 32, 32, 32, 32, 32, 32, 192, 3, 1, 1, 0);
    for (int j = 0; j < 3; ++j)
        resunit(32, 32, 32, 192, d_r[0][j][0], d_r[0][j][1], W_dr[0][j][0], W_dr[0][j][1]);
    convt(curB, W_up0, up0_b, s1, nullptr, 32, 32, 32, 192, 96, 0);
    { _Float16* t = curB; curB = s1; s1 = t; }
    for (int j = 0; j < 3; ++j)
        resunit(64, 64, 64, 96, d_r[1][j][0], d_r[1][j][1], W_dr[1][j][0], W_dr[1][j][1]);
    convt(curB, W_up1, up1_b, nullptr, (float*)d_out, 64, 64, 64, 96, 1, 1);
}
